// RelationshipAttention_77558519431550
// MI455X (gfx1250) — compile-verified, matrix operand order checked by reading, not run on hardware
//
#include <hip/hip_runtime.h>
#include <hip/hip_bf16.h>

typedef __attribute__((ext_vector_type(16))) _Float16 v16h;
typedef __attribute__((ext_vector_type(8)))  float    v8f;

#define BATCH 8
#define NN    2048
#define DD    256
#define KTOP  100
#define RTOP  5
#define BIGV  1000000000.0f
#define LNEPS 1e-5f

// ---------------------------------------------------------------------------
// Kernel 1: fp32 -> fp16 conversion of q and k (feeds WMMA)
// ---------------------------------------------------------------------------
__global__ void relattn_cvt_kernel(const float* __restrict__ q,
                                   const float* __restrict__ k,
                                   _Float16* __restrict__ qh,
                                   _Float16* __restrict__ kh, int n) {
    int idx = blockIdx.x * 256 + threadIdx.x;
    if (idx < n) {
        qh[idx] = (_Float16)q[idx];
        kh[idx] = (_Float16)k[idx];
    }
}

// ---------------------------------------------------------------------------
// Kernel 2: fused  scores1 = softmax(q @ k^T)  via v_wmma_f32_16x16x32_f16.
// One workgroup = 16 rows x 2048 cols of one batch. 8 wave32s, each wave owns
// column tiles jt = wave, wave+8, ... Logits staged in a 128KB LDS stripe
// (CDNA5: up to 320KB/WGP), softmax'd in LDS, written to HBM exactly once.
// Two independent accumulator chains (even/odd K-fragments) break the
// WMMA D->C RAW chain so independent WMMAs can issue back-to-back.
// ---------------------------------------------------------------------------
__global__ void relattn_gemm_softmax_kernel(const _Float16* __restrict__ qh,
                                            const _Float16* __restrict__ kh,
                                            float* __restrict__ out) {
    __shared__ float s_log[16][NN + 8];
    __shared__ float rbuf[16][17];

    const int tid  = threadIdx.x;
    const int b    = blockIdx.y;
    const int m0   = blockIdx.x * 16;
    const int wave = tid >> 5;
    const int lane = tid & 31;
    const int hl   = lane & 15;   // 0..15 : M (for A) / N (for B)
    const int hi   = lane >> 4;   // 0..1  : K-half select

    const _Float16* qb = qh + ((size_t)b * NN + m0) * DD;
    const _Float16* kb = kh + (size_t)b * NN * DD;

    // ---- A fragments: 16x256 f16 per the ISA 16-bit A 16x32 layout ----
    // lane L (row M=L%16); VGPR pair p: K = t*32 + (p<4?0:16) + (L>=16?8:0) + 2*(p%4)
    v16h afrag[8];
#pragma unroll
    for (int t = 0; t < 8; ++t) {
        const _Float16* qrow = qb + hl * DD + t * 32 + hi * 8;
        v16h a;
#pragma unroll
        for (int p = 0; p < 8; ++p) {
            int kk = ((p < 4) ? 0 : 16) + 2 * (p & 3);
            a[2 * p]     = qrow[kk];
            a[2 * p + 1] = qrow[kk + 1];
        }
        afrag[t] = a;
    }

    // ---- column tiles: each wave handles 16 of the 128 tiles ----
    for (int jt = wave; jt < NN / 16; jt += 8) {
        const int n0 = jt * 16;
        const _Float16* krow = kb + (size_t)(n0 + hl) * DD + hi * 8;

        // branch-free prefetch of next tile's k rows (global_prefetch_b8)
        const int pn0 = (jt + 8 < NN / 16) ? (n0 + 128) : n0;
        __builtin_prefetch(kb + (size_t)(pn0 + hl) * DD, 0, 1);

        // hoist all B fragments for this tile: 16 x global_load_b128 in flight
        v16h bfr[8];
#pragma unroll
        for (int t = 0; t < 8; ++t) {
            const _Float16* kp = krow + t * 32;
            v16h bf;
#pragma unroll
            for (int p = 0; p < 8; ++p) {
                int kk = ((p < 4) ? 0 : 16) + 2 * (p & 3);
                bf[2 * p]     = kp[kk];
                bf[2 * p + 1] = kp[kk + 1];
            }
            bfr[t] = bf;
        }

        // two independent WMMA chains (even / odd K-fragments)
        v8f c0 = {};
        v8f c1 = {};
#pragma unroll
        for (int t = 0; t < 4; ++t) {
            c0 = __builtin_amdgcn_wmma_f32_16x16x32_f16(
                    false, afrag[2 * t],     false, bfr[2 * t],     (short)0, c0, false, false);
            c1 = __builtin_amdgcn_wmma_f32_16x16x32_f16(
                    false, afrag[2 * t + 1], false, bfr[2 * t + 1], (short)0, c1, false, false);
        }

        // merge + scatter to LDS.  C/D layout: lane L, VGPR r ->
        //   M = r + (L>=16?8:0), N = L%16
#pragma unroll
        for (int r = 0; r < 8; ++r) {
            int M = r + hi * 8;
            s_log[M][n0 + hl] = c0[r] + c1[r];
        }
    }
    __syncthreads();

    // ---- row softmax over the 16x2048 LDS stripe (16 threads / row) ----
    const int row = tid >> 4;
    const int sub = tid & 15;

    float mx = -3.0e38f;
    for (int c2 = sub; c2 < NN; c2 += 16) mx = fmaxf(mx, s_log[row][c2]);
    rbuf[row][sub] = mx;
    __syncthreads();
    if (sub == 0) {
        float m = rbuf[row][0];
#pragma unroll
        for (int t2 = 1; t2 < 16; ++t2) m = fmaxf(m, rbuf[row][t2]);
        rbuf[row][16] = m;
    }
    __syncthreads();
    const float rowmax = rbuf[row][16];

    float sum = 0.0f;
    for (int c2 = sub; c2 < NN; c2 += 16) {
        float e = __expf(s_log[row][c2] - rowmax);
        s_log[row][c2] = e;
        sum += e;
    }
    rbuf[row][sub] = sum;
    __syncthreads();
    if (sub == 0) {
        float s2 = 0.0f;
#pragma unroll
        for (int t2 = 0; t2 < 16; ++t2) s2 += rbuf[row][t2];
        rbuf[row][16] = s2;
    }
    __syncthreads();
    const float inv = 1.0f / rbuf[row][16];

    float* orow = out + ((size_t)b * NN + m0 + row) * NN;
    for (int c2 = sub; c2 < NN; c2 += 16) orow[c2] = s_log[row][c2] * inv;
}

// ---------------------------------------------------------------------------
// Kernel 3: top-100 of diag(scores1), indices sorted ascending
// ---------------------------------------------------------------------------
__global__ void relattn_topk_kernel(const float* __restrict__ scores,
                                    int* __restrict__ top_idx) {
    __shared__ float vals[NN];
    __shared__ float lv[256];
    __shared__ int   li[256];
    __shared__ int   chosen[KTOP];

    const int b = blockIdx.x;
    const int tid = threadIdx.x;
    const float* sb = scores + (size_t)b * NN * NN;

    for (int i = tid; i < NN; i += 256) vals[i] = sb[(size_t)i * NN + i];
    __syncthreads();

    for (int it = 0; it < KTOP; ++it) {
        float best = -3.0e38f; int bidx = 0;
        for (int i = tid; i < NN; i += 256) {
            float v = vals[i];
            if (v > best) { best = v; bidx = i; }
        }
        lv[tid] = best; li[tid] = bidx;
        __syncthreads();
        for (int s = 128; s > 0; s >>= 1) {
            if (tid < s) {
                if (lv[tid + s] > lv[tid] ||
                    (lv[tid + s] == lv[tid] && li[tid + s] < li[tid])) {
                    lv[tid] = lv[tid + s]; li[tid] = li[tid + s];
                }
            }
            __syncthreads();
        }
        if (tid == 0) { chosen[it] = li[0]; vals[li[0]] = -3.0e38f; }
        __syncthreads();
    }

    if (tid == 0) {  // ascending insertion sort of 100 indices
        for (int a = 1; a < KTOP; ++a) {
            int v = chosen[a]; int c2 = a - 1;
            while (c2 >= 0 && chosen[c2] > v) { chosen[c2 + 1] = chosen[c2]; --c2; }
            chosen[c2 + 1] = v;
        }
    }
    __syncthreads();
    if (tid < KTOP) top_idx[b * KTOP + tid] = chosen[tid];
}

// ---------------------------------------------------------------------------
// Kernel 4: per (b,i): rel row gather, top-5 (diag=BIG), sort, emit soi + re
// ---------------------------------------------------------------------------
__global__ void relattn_rel_kernel(const float* __restrict__ scores,
                                   const float* __restrict__ q,
                                   const int* __restrict__ top_idx,
                                   float* __restrict__ soi,
                                   float* __restrict__ re) {
    __shared__ int   tix[KTOP];
    __shared__ float relv[KTOP];
    __shared__ int   rsel[RTOP];
    __shared__ float red[256];

    const int b = blockIdx.y, i = blockIdx.x, tid = threadIdx.x;

    if (tid < KTOP) tix[tid] = top_idx[b * KTOP + tid];
    __syncthreads();
    const int rowi = tix[i];

    if (tid < KTOP)
        relv[tid] = (tid == i) ? BIGV
                               : scores[((size_t)b * NN + rowi) * NN + tix[tid]];
    __syncthreads();

    if (tid == 0) {
        bool used[KTOP];
        for (int j = 0; j < KTOP; ++j) used[j] = false;
        int sel[RTOP];
        for (int r = 0; r < RTOP; ++r) {
            float best = -3.0e38f; int bj = 0;
            for (int j = 0; j < KTOP; ++j)
                if (!used[j] && relv[j] > best) { best = relv[j]; bj = j; }
            used[bj] = true; sel[r] = bj;
        }
        for (int a = 1; a < RTOP; ++a) {  // ascending
            int v = sel[a]; int c2 = a - 1;
            while (c2 >= 0 && sel[c2] > v) { sel[c2 + 1] = sel[c2]; --c2; }
            sel[c2 + 1] = v;
        }
        for (int r = 0; r < RTOP; ++r) rsel[r] = sel[r];
    }
    __syncthreads();

    const int subj = rowi;
    for (int r = 0; r < RTOP; ++r) {
        const int obj = tix[rsel[r]];
        const size_t e = ((size_t)b * KTOP + i) * RTOP + r;
        if (tid == 0) {
            soi[e * 3 + 0] = (float)b;
            soi[e * 3 + 1] = (float)subj;
            soi[e * 3 + 2] = (float)obj;
        }
        float v = q[((size_t)b * NN + subj) * DD + tid] +
                  q[((size_t)b * NN + obj)  * DD + tid];
        red[tid] = v; __syncthreads();
        for (int s = 128; s > 0; s >>= 1) {
            if (tid < s) red[tid] += red[tid + s];
            __syncthreads();
        }
        const float mean = red[0] / (float)DD;
        __syncthreads();
        red[tid] = v * v; __syncthreads();
        for (int s = 128; s > 0; s >>= 1) {
            if (tid < s) red[tid] += red[tid + s];
            __syncthreads();
        }
        const float var = red[0] / (float)DD - mean * mean;
        __syncthreads();
        re[e * DD + tid] = (v - mean) * rsqrtf(var + LNEPS);
    }
}

// ---------------------------------------------------------------------------
extern "C" void kernel_launch(void* const* d_in, const int* in_sizes, int n_in,
                              void* d_out, int out_size, void* d_ws, size_t ws_size,
                              hipStream_t stream) {
    const float* q = (const float*)d_in[0];
    const float* k = (const float*)d_in[1];
    // d_in[2]/d_in[3]: top_k scalars (compile-time constants 100 / 5 here)

    // workspace layout
    const size_t nqk = (size_t)BATCH * NN * DD;      // 4,194,304 elems
    _Float16* qh = (_Float16*)d_ws;
    _Float16* kh = qh + nqk;
    int* top_idx = (int*)(kh + nqk);

    // output layout: scores1 | soi | re  (flattened tuple)
    float* scores = (float*)d_out;
    float* soi = scores + (size_t)BATCH * NN * NN;
    float* re  = soi + (size_t)BATCH * KTOP * RTOP * 3;

    relattn_cvt_kernel<<<(int)((nqk + 255) / 256), 256, 0, stream>>>(q, k, qh, kh, (int)nqk);

    dim3 gg(NN / 16, BATCH);
    relattn_gemm_softmax_kernel<<<gg, 256, 0, stream>>>(qh, kh, scores);

    relattn_topk_kernel<<<BATCH, 256, 0, stream>>>(scores, top_idx);

    dim3 gr(KTOP, BATCH);
    relattn_rel_kernel<<<gr, 256, 0, stream>>>(scores, q, top_idx, soi, re);
}
